// Head_46213848104973
// MI455X (gfx1250) — matrix operand order repaired and checked by reading, not compiled, not run on hardware
//
#include <hip/hip_runtime.h>
#include <math.h>

// CDNA5 wave32 WMMA types
typedef _Float16 h16;
typedef __attribute__((ext_vector_type(16))) _Float16 v16h;
typedef __attribute__((ext_vector_type(8)))  _Float16 v8h;
typedef __attribute__((ext_vector_type(8)))  float    v8f;

#define BB 8
#define TT 2048
#define CC 1024
#define HH 64
#define BT (BB * TT)   // 16384 rows total
#define WN (CC * HH)   // 65536 elements per weight matrix

// ---------------------------------------------------------------------------
// Phase 0: one-time f32 -> f16 conversion of the three weight matrices so the
// GEMM B-operand is 16 direct strided f16 loads (no per-k-step cvt VALU).
// Wh layout: [m][k][n], m in {k,q,v}.
// ---------------------------------------------------------------------------
__global__ __launch_bounds__(256) void convert_w(
    const float* __restrict__ Wk, const float* __restrict__ Wq,
    const float* __restrict__ Wv, h16* __restrict__ Wh)
{
    const int idx = blockIdx.x * 256 + threadIdx.x;
    if (idx < WN) {
        Wh[idx]          = (h16)Wk[idx];
        Wh[WN + idx]     = (h16)Wq[idx];
        Wh[2 * WN + idx] = (h16)Wv[idx];
    }
}

// ---------------------------------------------------------------------------
// Phase 1: fused QKV projection.  One wave per 16-row tile computes the
// 16x64 output tile of K, Q AND V (12 f32 accumulators), reading x once.
//   A layout (16x32 f16): lane L<16 holds row L, K = {kk..kk+7, kk+16..kk+23};
//                         lane L>=16 holds row L-16, K = {kk+8..15, kk+24..31}
//   B layout (32x16 f16): lane L holds col L%16, K same as A for its half.
// k is stored TRANSPOSED (kT[b][h][t]) so phase 2's S B-operand loads are
// contiguous.  q is pre-scaled by C^-0.5.
// ---------------------------------------------------------------------------
__global__ __launch_bounds__(128) void qkv_proj(
    const float* __restrict__ x, const h16* __restrict__ Wh,
    const float* __restrict__ bk, const float* __restrict__ bq,
    const float* __restrict__ bv,
    h16* __restrict__ qb, h16* __restrict__ kTb, h16* __restrict__ vbuf)
{
    const int wave    = threadIdx.x >> 5;
    const int lane    = threadIdx.x & 31;
    const int rowtile = blockIdx.x * 4 + wave;     // 0..1023 (16-row tiles of BT)
    const int halfsel = lane >> 4;
    const int l16     = lane & 15;

    const int arow = rowtile * 16 + l16;           // global row this lane owns (A)
    const float* xr = x + (size_t)arow * CC;

    v8f acc[3][4] = {};                            // {k,q,v} x 4 N-tiles

    for (int kk = 0; kk < CC; kk += 32) {
        // ---- A operand: convert 16 f32 -> v16h in WMMA A layout ----
        const int base = kk + halfsel * 8;
        const float4 f0 = *(const float4*)(xr + base);
        const float4 f1 = *(const float4*)(xr + base + 4);
        const float4 f2 = *(const float4*)(xr + base + 16);
        const float4 f3 = *(const float4*)(xr + base + 20);
        v16h a;
        a[0]=(h16)f0.x; a[1]=(h16)f0.y; a[2]=(h16)f0.z; a[3]=(h16)f0.w;
        a[4]=(h16)f1.x; a[5]=(h16)f1.y; a[6]=(h16)f1.z; a[7]=(h16)f1.w;
        a[8]=(h16)f2.x; a[9]=(h16)f2.y; a[10]=(h16)f2.z; a[11]=(h16)f2.w;
        a[12]=(h16)f3.x; a[13]=(h16)f3.y; a[14]=(h16)f3.z; a[15]=(h16)f3.w;

        // ---- B operands: direct f16 loads, 12 independent WMMAs on one A ----
        const h16* wr = Wh + (size_t)(kk + halfsel * 8) * HH + l16;
        #pragma unroll
        for (int m = 0; m < 3; ++m) {
            #pragma unroll
            for (int nt = 0; nt < 4; ++nt) {
                const h16* wp = wr + (size_t)m * WN + nt * 16;
                v16h bm;
                #pragma unroll
                for (int e = 0; e < 8; ++e) { bm[e] = wp[(size_t)e * HH]; bm[8 + e] = wp[(size_t)(16 + e) * HH]; }
                acc[m][nt] = __builtin_amdgcn_wmma_f32_16x16x32_f16(
                    false, a, false, bm, (short)0, acc[m][nt], false, false);
            }
        }
    }

    // ---- epilogue: bias, q-scale, store f16 (kT transposed) ----
    const float qscale = rsqrtf((float)CC);
    #pragma unroll
    for (int m = 0; m < 3; ++m) {
        const float* bias = (m == 0) ? bk : (m == 1) ? bq : bv;
        #pragma unroll
        for (int nt = 0; nt < 4; ++nt) {
            #pragma unroll
            for (int i = 0; i < 8; ++i) {
                const int r    = i + halfsel * 8;      // C/D layout: row
                const int grow = rowtile * 16 + r;
                const int n    = nt * 16 + l16;        // C/D layout: col
                float val = acc[m][nt][i] + bias[n];
                if (m == 0) {
                    const int bidx = grow / TT, t = grow % TT;
                    kTb[((size_t)bidx * HH + n) * TT + t] = (h16)val;  // transposed
                } else if (m == 1) {
                    qb[(size_t)grow * HH + n] = (h16)(val * qscale);
                } else {
                    vbuf[(size_t)grow * HH + n] = (h16)val;
                }
            }
        }
    }
}

// ---------------------------------------------------------------------------
// Phase 2: causal flash attention. One wave owns one 16-query tile.
// S/PV via wmma_f32_16x16x32_f16; online softmax on C/D layout;
// P converted C/D->A layout through padded LDS.
// qtile = wave*32 + qgrp balances causal trip counts across blocks.
// ---------------------------------------------------------------------------
__global__ __launch_bounds__(128) void flash_attn(
    const h16* __restrict__ qb, const h16* __restrict__ kTb,
    const h16* __restrict__ vbuf, float* __restrict__ out)
{
    __shared__ h16 pbuf[4][16][40];                // +8 halfs row pad (80B stride)

    const int wave    = threadIdx.x >> 5;
    const int lane    = threadIdx.x & 31;
    const int b       = blockIdx.x >> 5;           // 0..7
    const int qgrp    = blockIdx.x & 31;           // 0..31
    const int qtile   = wave * 32 + qgrp;          // 0..127, balanced per block
    const int halfsel = lane >> 4;
    const int l16     = lane & 15;

    // ---- Q tile in A layout, split into feature halves (K=0..31, 32..63) ----
    const h16* qr = qb + ((size_t)b * TT + qtile * 16 + l16) * HH;
    v16h qa[2];
    #pragma unroll
    for (int fh = 0; fh < 2; ++fh) {
        const int base = fh * 32 + halfsel * 8;
        const v8h c0 = *(const v8h*)(qr + base);
        const v8h c1 = *(const v8h*)(qr + base + 16);
        #pragma unroll
        for (int e = 0; e < 8; ++e) { qa[fh][e] = c0[e]; qa[fh][8 + e] = c1[e]; }
    }

    v8f o[4] = {};
    float mrow[8], lrow[8];
    #pragma unroll
    for (int i = 0; i < 8; ++i) { mrow[i] = -INFINITY; lrow[i] = 0.0f; }

    const int nkb = qtile / 2 + 1;                 // 32-key blocks (causal bound)
    for (int kbi = 0; kbi < nkb; ++kbi) {
        const int kb = kbi * 32;
        const bool need_mask = (kb + 32) > qtile * 16;

        if (kbi + 1 < nkb) {                        // global_prefetch_b8
            __builtin_prefetch(kTb + ((size_t)b * HH + lane) * TT + kb + 32, 0, 0);
            __builtin_prefetch(vbuf + ((size_t)b * TT + kb + 32 + lane) * HH, 0, 0);
        }

        // ---- S = Q . K^T : two 16-key sub-tiles, K accumulated over 64 ----
        v8f s[2];
        #pragma unroll
        for (int sI = 0; sI < 2; ++sI) {
            v8f z = {};
            #pragma unroll
            for (int fh = 0; fh < 2; ++fh) {
                const h16* kp = kTb + ((size_t)b * HH + fh * 32 + halfsel * 8) * TT + kb + sI * 16 + l16;
                v16h kbm;   // col = key l16, K = features (table order)
                #pragma unroll
                for (int e = 0; e < 8; ++e) { kbm[e] = kp[(size_t)e * TT]; kbm[8 + e] = kp[(size_t)(16 + e) * TT]; }
                z = __builtin_amdgcn_wmma_f32_16x16x32_f16(
                    false, qa[fh], false, kbm, (short)0, z, false, false);
            }
            s[sI] = z;
        }

        // ---- online softmax over the 16x32 block (row = acc index + half) ----
        #pragma unroll
        for (int i = 0; i < 8; ++i) {
            float v0 = s[0][i], v1 = s[1][i];
            if (need_mask) {
                const int row = qtile * 16 + i + halfsel * 8;
                if (kb + l16 > row)      v0 = -INFINITY;
                if (kb + 16 + l16 > row) v1 = -INFINITY;
            }
            float mx = fmaxf(v0, v1);
            mx = fmaxf(mx, __shfl_xor(mx, 1));
            mx = fmaxf(mx, __shfl_xor(mx, 2));
            mx = fmaxf(mx, __shfl_xor(mx, 4));
            mx = fmaxf(mx, __shfl_xor(mx, 8));
            const float newm = fmaxf(mrow[i], mx);
            const float p0 = __expf(v0 - newm);
            const float p1 = __expf(v1 - newm);
            float rs = p0 + p1;
            rs += __shfl_xor(rs, 1);
            rs += __shfl_xor(rs, 2);
            rs += __shfl_xor(rs, 4);
            rs += __shfl_xor(rs, 8);
            const float corr = __expf(mrow[i] - newm);
            lrow[i] = lrow[i] * corr + rs;
            mrow[i] = newm;
            o[0][i] *= corr; o[1][i] *= corr; o[2][i] *= corr; o[3][i] *= corr;

            const int r = i + halfsel * 8;
            pbuf[wave][r][l16]      = (h16)p0;     // C/D layout -> LDS
            pbuf[wave][r][16 + l16] = (h16)p1;
        }
        asm volatile("s_wait_dscnt 0" ::: "memory");

        // ---- P back in A layout (16x32) ----
        v16h pa;
        {
            const int base = halfsel * 8;
            const v8h c0 = *(const v8h*)(&pbuf[wave][l16][base]);
            const v8h c1 = *(const v8h*)(&pbuf[wave][l16][base + 16]);
            #pragma unroll
            for (int e = 0; e < 8; ++e) { pa[e] = c0[e]; pa[8 + e] = c1[e]; }
        }

        // ---- O += P . V : B-operand lane holds col l16, keys in table order ----
        const h16* vr = vbuf + ((size_t)b * TT + kb + halfsel * 8) * HH + l16;
        #pragma unroll
        for (int nt = 0; nt < 4; ++nt) {
            v16h vbm;
            #pragma unroll
            for (int e = 0; e < 8; ++e) { vbm[e] = vr[nt * 16 + e * HH]; vbm[8 + e] = vr[nt * 16 + (16 + e) * HH]; }
            o[nt] = __builtin_amdgcn_wmma_f32_16x16x32_f16(
                false, pa, false, vbm, (short)0, o[nt], false, false);
        }
    }

    // ---- epilogue: out = O / l, f32 ----
    #pragma unroll
    for (int nt = 0; nt < 4; ++nt) {
        #pragma unroll
        for (int i = 0; i < 8; ++i) {
            const int r = i + halfsel * 8;
            out[((size_t)b * TT + qtile * 16 + r) * HH + nt * 16 + l16] =
                o[nt][i] / lrow[i];
        }
    }
}

// ---------------------------------------------------------------------------
extern "C" void kernel_launch(void* const* d_in, const int* in_sizes, int n_in,
                              void* d_out, int out_size, void* d_ws, size_t ws_size,
                              hipStream_t stream) {
    (void)in_sizes; (void)n_in; (void)out_size; (void)ws_size;

    const float* x  = (const float*)d_in[0];
    const float* Wk = (const float*)d_in[1];
    const float* bk = (const float*)d_in[2];
    const float* Wq = (const float*)d_in[3];
    const float* bq = (const float*)d_in[4];
    const float* Wv = (const float*)d_in[5];
    const float* bv = (const float*)d_in[6];
    float* out = (float*)d_out;

    // workspace: q, kT, v as f16 (2 MB each) + Wh f16 (384 KB)
    h16* qb  = (h16*)d_ws;
    h16* kTb = qb  + (size_t)BT * HH;
    h16* vb  = kTb + (size_t)BT * HH;
    h16* Wh  = vb  + (size_t)BT * HH;

    convert_w<<<WN / 256, 256, 0, stream>>>(Wk, Wq, Wv, Wh);

    dim3 grid1(BT / 16 / 4);             // 256 blocks, 4 waves each, fused KQV
    qkv_proj<<<grid1, 128, 0, stream>>>(x, Wh, bk, bq, bv, qb, kTb, vb);

    dim3 grid2(BB * (TT / 16) / 4);      // 256 blocks, 4 waves each
    flash_attn<<<grid2, 128, 0, stream>>>(qb, kTb, vb, out);
}
